// Transformer_40149354283328
// MI455X (gfx1250) — matrix operand order repaired and checked by reading, not compiled, not run on hardware
//
#include <hip/hip_runtime.h>
#include <hip/hip_bf16.h>

typedef __bf16 bf16_t;
typedef __attribute__((ext_vector_type(16))) __bf16 v16bf;
typedef __attribute__((ext_vector_type(8)))  float  v8f;

// Problem constants (match reference)
enum : int { LYR = 6, CH = 384, NH = 4, RW = 4, DKH = 96, CCH = 1536, KS = 3,
             NB = 16, TT = 512 };
#define NT (NB * TT)            // 8192 tokens
#define TP (TT + 2)             // padded sequence length for K=3 convs

// gemm flags
enum : int {
    F_STORE_T  = 1,   // store D transposed: D[n*ldd + m]  (token-major out)
    F_BIAS     = 2,
    F_RELU     = 4,
    F_ACC      = 8,
    F_OUT_F32  = 16,
    F_OUT_BF16 = 32,
    F_MASKN    = 64,  // multiply output column n by nmask[z*sMz + n]
};

static inline int cdiv(int a, int b) { return (a + b - 1) / b; }

// ---------------------------------------------------------------------------
// Generic bf16 WMMA GEMM:  D[M,N] = alpha*(A[M,K(,taps)] x B + bias).
//   A  row-major [M, taps*Kd]  (per row: [tap][k])
//   Bm "Bt" layout [N, K]: row n holds the contraction dim contiguously;
//      tap shifts the B row pointer by tap*ldb (padded conv input rows).
// Each wave computes a 32x32 region = 2x2 WMMA tiles (4 independent
// accumulator chains, 2 b128 loads per v_wmma_f32_16x16x32_bf16).
// Block = 256 threads = 8 waves arranged 2(M) x 4(N) -> 64x128 block tile.
// All M,N used here are multiples of 32, so tile guards stay wave-uniform
// and EXEC is all-ones for every WMMA.
// ---------------------------------------------------------------------------
__global__ __launch_bounds__(256) void gemm_wmma(
    const bf16_t* __restrict__ A, int lda, long sAz,
    const bf16_t* __restrict__ Bm, int ldb, long sBz,
    float* __restrict__ Df, long sDfz,
    bf16_t* __restrict__ Db, long sDbz,
    int ldd, int M, int N, int Kd, int taps,
    const float* __restrict__ bias, float alpha, int flags,
    const float* __restrict__ nmask, long sMz)
{
    const int z    = blockIdx.z;
    const int wave = threadIdx.x >> 5;
    const int lane = threadIdx.x & 31;
    const int m0 = blockIdx.y * 64 + (wave >> 2) * 32;
    const int n0 = blockIdx.x * 128 + (wave & 3) * 32;
    if (m0 >= M || n0 >= N) return;           // wave-uniform: EXEC stays all-1s

    const int l16 = lane & 15;
    const int ah  = (lane < 16) ? 0 : 8;      // A kbase per ISA 16-bit A layout
    const int bh  = (lane < 16) ? 0 : 8;      // B kbase, same k order as A
    const bf16_t* arow0 = A + sAz * z + (long)(m0 + l16) * lda;
    const bf16_t* arow1 = arow0 + (long)16 * lda;
    const bf16_t* brow0 = Bm + sBz * z + (long)(n0 + l16) * ldb;
    const bf16_t* brow1 = brow0 + (long)16 * ldb;

    v8f acc00 = {}, acc01 = {}, acc10 = {}, acc11 = {};
    for (int tap = 0; tap < taps; ++tap) {
        const long aoff = (long)tap * Kd;     // A: [m][tap][k]
        const long boff = (long)tap * ldb;    // B: shift one (padded) row per tap
        for (int k = 0; k < Kd; k += 32) {
            union { uint4 u[2]; v16bf v; } ua0, ua1, ub0, ub1;
            // A lane: row m=lane&15, k chunks [kbase,kbase+8) and [kbase+16,kbase+24)
            ua0.u[0] = *(const uint4*)(arow0 + aoff + k + ah);
            ua0.u[1] = *(const uint4*)(arow0 + aoff + k + ah + 16);
            ua1.u[0] = *(const uint4*)(arow1 + aoff + k + ah);
            ua1.u[1] = *(const uint4*)(arow1 + aoff + k + ah + 16);
            // B lane: col n=lane&15, k chunks [kbase,kbase+8) and [kbase+16,kbase+24)
            ub0.u[0] = *(const uint4*)(brow0 + boff + k + bh);
            ub0.u[1] = *(const uint4*)(brow0 + boff + k + bh + 16);
            ub1.u[0] = *(const uint4*)(brow1 + boff + k + bh);
            ub1.u[1] = *(const uint4*)(brow1 + boff + k + bh + 16);
            acc00 = __builtin_amdgcn_wmma_f32_16x16x32_bf16(
                false, ua0.v, false, ub0.v, (short)0, acc00, false, false);
            acc01 = __builtin_amdgcn_wmma_f32_16x16x32_bf16(
                false, ua0.v, false, ub1.v, (short)0, acc01, false, false);
            acc10 = __builtin_amdgcn_wmma_f32_16x16x32_bf16(
                false, ua1.v, false, ub0.v, (short)0, acc10, false, false);
            acc11 = __builtin_amdgcn_wmma_f32_16x16x32_bf16(
                false, ua1.v, false, ub1.v, (short)0, acc11, false, false);
        }
    }

    // D tile layout: lanes 0-15 -> N=lane, M=r ; lanes 16-31 -> N=lane-16, M=8+r
    const int mb8 = (lane >> 4) << 3;
#pragma unroll
    for (int tm = 0; tm < 2; ++tm) {
        const int mt = m0 + tm * 16;
        if (mt >= M) continue;
#pragma unroll
        for (int tn = 0; tn < 2; ++tn) {
            const int nt = n0 + tn * 16;
            if (nt >= N) continue;
            const v8f acc = (tm == 0) ? (tn == 0 ? acc00 : acc01)
                                      : (tn == 0 ? acc10 : acc11);
            const int n = nt + l16;
            float mskv = 1.0f;
            if (flags & F_MASKN) mskv = nmask[sMz * z + n];
#pragma unroll
            for (int r = 0; r < 8; ++r) {
                const int m = mt + mb8 + r;
                float v = acc[r];
                if (flags & F_BIAS) v += bias[m];
                v *= alpha;
                const long idx = (flags & F_STORE_T) ? ((long)n * ldd + m)
                                                     : ((long)m * ldd + n);
                if (flags & F_ACC)  v += Df[sDfz * z + idx];
                if (flags & F_RELU) v = fmaxf(v, 0.0f);
                v *= mskv;
                if (flags & F_OUT_F32)  Df[sDfz * z + idx] = v;
                if (flags & F_OUT_BF16) Db[sDbz * z + idx] = (bf16_t)v;
            }
        }
    }
}

// ---------------------------------------------------------------------------
// Elementwise helpers
// ---------------------------------------------------------------------------
__global__ void cvt_f32_bf16(const float* __restrict__ s, bf16_t* __restrict__ d, long n) {
    long i = (long)blockIdx.x * 256 + threadIdx.x;
    if (i < n) d[i] = (bf16_t)s[i];
}
__global__ void zero_bf16(bf16_t* __restrict__ d, long n) {
    long i = (long)blockIdx.x * 256 + threadIdx.x;
    if (i < n) d[i] = (bf16_t)0.0f;
}
// conv weight repack: w[row][c][k] (k fastest) -> d[row][k][c]   (row = l*Cout+m)
__global__ void repack_conv(const float* __restrict__ w, bf16_t* __restrict__ d,
                            int Cin, long n) {
    long i = (long)blockIdx.x * 256 + threadIdx.x;
    if (i >= n) return;
    int  c   = (int)(i % Cin);
    long r   = i / Cin;
    int  k   = (int)(r % KS);
    long row = r / KS;
    d[i] = (bf16_t)w[(row * Cin + c) * KS + k];
}
// x [B,C,T] -> token-major [B*T, C], masked, f32 + bf16
__global__ void transpose_in(const float* __restrict__ x, const float* __restrict__ mask,
                             float* __restrict__ Xf, bf16_t* __restrict__ Xb) {
    long i = (long)blockIdx.x * 256 + threadIdx.x;
    if (i >= (long)NB * CH * TT) return;
    int t = (int)(i % TT);
    int c = (int)((i / TT) % CH);
    int b = (int)(i / ((long)TT * CH));
    float v = x[i] * mask[b * TT + t];
    long  o = ((long)(b * TT + t)) * CH + c;
    Xf[o] = v; Xb[o] = (bf16_t)v;
}
// token-major [B*T, C] -> out [B,C,T], masked
__global__ void transpose_out(const float* __restrict__ Xf, const float* __restrict__ mask,
                              float* __restrict__ out) {
    long i = (long)blockIdx.x * 256 + threadIdx.x;
    if (i >= (long)NB * CH * TT) return;
    int t = (int)(i % TT);
    int c = (int)((i / TT) % CH);
    int b = (int)(i / ((long)TT * CH));
    out[i] = Xf[((long)(b * TT + t)) * CH + c] * mask[b * TT + t];
}

// ---------------------------------------------------------------------------
// Banded rel-K add + mask + softmax.  One 128-thread block per (t, h) row,
// launched per batch b.  S f32 [H,T,T] -> P bf16 [H,T,T].
// scores[t,s] += qs[t,:] . rel_k[(s-t)+RW,:]   for |s-t| <= RW
// ---------------------------------------------------------------------------
__global__ __launch_bounds__(128) void softmax_rel(
    const float* __restrict__ S, bf16_t* __restrict__ P,
    const bf16_t* __restrict__ Qbf, const float* __restrict__ relk,
    const float* __restrict__ mask, int b)
{
    const int t = blockIdx.x, h = blockIdx.y, tid = threadIdx.x;
    __shared__ float band[2 * RW + 1];
    __shared__ float red[128];

    if (tid < 2 * RW + 1) {
        int s = t + tid - RW;
        float v = 0.0f;
        if (s >= 0 && s < TT) {
            const bf16_t* q  = Qbf + ((long)(b * TT + t)) * CH + h * DKH;  // qs (scale folded)
            const float*  rk = relk + tid * DKH;
            for (int d = 0; d < DKH; ++d) v += (float)q[d] * rk[d];
        }
        band[tid] = v;
    }
    __syncthreads();

    const float* srow = S + ((long)h * TT + t) * TT;
    const float  mrow = mask[b * TT + t];
    float vals[4];
    float mx = -3.0e38f;
#pragma unroll
    for (int i = 0; i < 4; ++i) {
        int s = tid + i * 128;
        float v = srow[s];
        int d = s - t;
        if (d >= -RW && d <= RW) v += band[d + RW];
        float mk = mrow * mask[b * TT + s];
        v = (mk > 0.0f) ? v : -1.0e4f;
        vals[i] = v; mx = fmaxf(mx, v);
    }
    red[tid] = mx; __syncthreads();
    for (int o = 64; o > 0; o >>= 1) { if (tid < o) red[tid] = fmaxf(red[tid], red[tid + o]); __syncthreads(); }
    mx = red[0]; __syncthreads();

    float sum = 0.0f;
#pragma unroll
    for (int i = 0; i < 4; ++i) { vals[i] = __expf(vals[i] - mx); sum += vals[i]; }
    red[tid] = sum; __syncthreads();
    for (int o = 64; o > 0; o >>= 1) { if (tid < o) red[tid] += red[tid + o]; __syncthreads(); }
    const float inv = 1.0f / red[0];

    bf16_t* prow = P + ((long)h * TT + t) * TT;
#pragma unroll
    for (int i = 0; i < 4; ++i) prow[tid + i * 128] = (bf16_t)(vals[i] * inv);
}

// out[t,d] += sum_{|δ|<=RW} p[t,t+δ] * rel_v[δ+RW, d]   (per b,h)
__global__ __launch_bounds__(DKH) void relv_add(
    const bf16_t* __restrict__ P, const float* __restrict__ relv,
    float* __restrict__ attnF, int b)
{
    const int t = blockIdx.x, h = blockIdx.y, d = threadIdx.x;
    const bf16_t* prow = P + ((long)h * TT + t) * TT;
    float acc = 0.0f;
#pragma unroll
    for (int j = 0; j < 2 * RW + 1; ++j) {
        int s = t + j - RW;
        if (s >= 0 && s < TT) acc += (float)prow[s] * relv[j * DKH + d];
    }
    attnF[((long)(b * TT + t)) * CH + h * DKH + d] += acc;
}

// ---------------------------------------------------------------------------
// LayerNorm over C for each token:  out = LN(maskY?Y*m:Y + R) * gamma + beta
// Writes f32 (optionally masked) and bf16 flat / padded-conv-layout copies.
// ---------------------------------------------------------------------------
__global__ __launch_bounds__(128) void ln_kernel(
    const float* __restrict__ Y, const float* __restrict__ R,
    const float* __restrict__ g, const float* __restrict__ be,
    const float* __restrict__ mask, int maskY,
    float* __restrict__ Xf, int maskXf,
    bf16_t* __restrict__ XbFlat, bf16_t* __restrict__ XbPad, int maskXb)
{
    __shared__ float red[128];
    const int tok = blockIdx.x;
    const int bb = tok / TT, t = tok % TT;
    const float mk = mask[bb * TT + t];
    const int tid = threadIdx.x;

    float s[3]; float sum = 0.0f;
#pragma unroll
    for (int i = 0; i < 3; ++i) {
        int c = tid + i * 128;
        float y = Y[(long)tok * CH + c];
        if (maskY) y *= mk;
        s[i] = y + R[(long)tok * CH + c];
        sum += s[i];
    }
    red[tid] = sum; __syncthreads();
    for (int o = 64; o > 0; o >>= 1) { if (tid < o) red[tid] += red[tid + o]; __syncthreads(); }
    const float mean = red[0] * (1.0f / CH); __syncthreads();

    float vs = 0.0f;
#pragma unroll
    for (int i = 0; i < 3; ++i) { float d = s[i] - mean; vs += d * d; }
    red[tid] = vs; __syncthreads();
    for (int o = 64; o > 0; o >>= 1) { if (tid < o) red[tid] += red[tid + o]; __syncthreads(); }
    const float inv = rsqrtf(red[0] * (1.0f / CH) + 1.0e-4f);

#pragma unroll
    for (int i = 0; i < 3; ++i) {
        int c = tid + i * 128;
        float o_ = (s[i] - mean) * inv * g[c] + be[c];
        if (Xf) Xf[(long)tok * CH + c] = maskXf ? o_ * mk : o_;
        float ob = maskXb ? o_ * mk : o_;
        if (XbFlat) XbFlat[(long)tok * CH + c] = (bf16_t)ob;
        if (XbPad)  XbPad[((long)bb * TP + 1 + t) * CH + c] = (bf16_t)ob;
    }
}

// ---------------------------------------------------------------------------
extern "C" void kernel_launch(void* const* d_in, const int* in_sizes, int n_in,
                              void* d_out, int out_size, void* d_ws, size_t ws_size,
                              hipStream_t stream)
{
    const float* x    = (const float*)d_in[0];
    const float* mask = (const float*)d_in[1];
    const float* Wq   = (const float*)d_in[2];
    const float* bq   = (const float*)d_in[3];
    const float* Wk   = (const float*)d_in[4];
    const float* bk   = (const float*)d_in[5];
    const float* Wv   = (const float*)d_in[6];
    const float* bv   = (const float*)d_in[7];
    const float* Wo   = (const float*)d_in[8];
    const float* bo   = (const float*)d_in[9];
    const float* relk = (const float*)d_in[10];
    const float* relv = (const float*)d_in[11];
    const float* ln0g = (const float*)d_in[12];
    const float* ln0b = (const float*)d_in[13];
    const float* c0w  = (const float*)d_in[14];
    const float* c0b  = (const float*)d_in[15];
    const float* c1w  = (const float*)d_in[16];
    const float* c1b  = (const float*)d_in[17];
    const float* ln1g = (const float*)d_in[18];
    const float* ln1b = (const float*)d_in[19];

    char* ws = (char*)d_ws;
    auto alloc = [&](size_t bytes) -> void* {
        void* p = (void*)ws;
        ws += (bytes + 255) & ~(size_t)255;
        return p;
    };
    // bf16 weights
    bf16_t* Wqb = (bf16_t*)alloc((size_t)LYR * CH * CH * 2);
    bf16_t* Wkb = (bf16_t*)alloc((size_t)LYR * CH * CH * 2);
    bf16_t* Wvb = (bf16_t*)alloc((size_t)LYR * CH * CH * 2);
    bf16_t* Wob = (bf16_t*)alloc((size_t)LYR * CH * CH * 2);
    bf16_t* W0p = (bf16_t*)alloc((size_t)LYR * CCH * KS * CH * 2);
    bf16_t* W1p = (bf16_t*)alloc((size_t)LYR * CH * KS * CCH * 2);
    // activations
    float*  Xf    = (float*) alloc((size_t)NT * CH * 4);
    float*  newX  = (float*) alloc((size_t)NT * CH * 4);
    bf16_t* Xbf   = (bf16_t*)alloc((size_t)NT * CH * 2);
    bf16_t* Xpad  = (bf16_t*)alloc((size_t)NB * TP * CH * 2);
    bf16_t* Qbf   = (bf16_t*)alloc((size_t)NT * CH * 2);
    bf16_t* Kbf   = (bf16_t*)alloc((size_t)NT * CH * 2);
    bf16_t* Vbf   = (bf16_t*)alloc((size_t)CH * NT * 2);   // channel-major
    float*  S     = (float*) alloc((size_t)NH * TT * TT * 4);
    bf16_t* P     = (bf16_t*)alloc((size_t)NH * TT * TT * 2);
    float*  attnF = (float*) alloc((size_t)NT * CH * 4);
    bf16_t* attnB = (bf16_t*)alloc((size_t)NT * CH * 2);
    float*  tmpF  = (float*) alloc((size_t)NT * CH * 4);
    bf16_t* Y0pad = (bf16_t*)alloc((size_t)NB * TP * CCH * 2);
    float*  Y1    = (float*) alloc((size_t)NT * CH * 4);
    (void)ws_size; (void)in_sizes; (void)n_in; (void)out_size;

    auto gemm = [&](const bf16_t* A, int lda, long sAz,
                    const bf16_t* Bm, int ldb, long sBz,
                    float* Df, long sDfz, bf16_t* Db, long sDbz, int ldd,
                    int M, int N, int Kd, int taps,
                    const float* bias, float alpha, int flags,
                    const float* nmask, long sMz, int Z) {
        dim3 g(cdiv(N, 128), cdiv(M, 64), Z);
        gemm_wmma<<<g, 256, 0, stream>>>(A, lda, sAz, Bm, ldb, sBz,
                                         Df, sDfz, Db, sDbz, ldd, M, N, Kd, taps,
                                         bias, alpha, flags, nmask, sMz);
    };

    // --- setup: weight conversion / repack, zero padded buffers, transpose in ---
    {
        long n = (long)LYR * CH * CH;
        cvt_f32_bf16<<<cdiv((int)n, 256), 256, 0, stream>>>(Wq, Wqb, n);
        cvt_f32_bf16<<<cdiv((int)n, 256), 256, 0, stream>>>(Wk, Wkb, n);
        cvt_f32_bf16<<<cdiv((int)n, 256), 256, 0, stream>>>(Wv, Wvb, n);
        cvt_f32_bf16<<<cdiv((int)n, 256), 256, 0, stream>>>(Wo, Wob, n);
        long n0 = (long)LYR * CCH * KS * CH;
        repack_conv<<<cdiv((int)n0, 256), 256, 0, stream>>>(c0w, W0p, CH, n0);
        long n1 = (long)LYR * CH * KS * CCH;
        repack_conv<<<cdiv((int)n1, 256), 256, 0, stream>>>(c1w, W1p, CCH, n1);
        long nxp = (long)NB * TP * CH;
        zero_bf16<<<cdiv((int)nxp, 256), 256, 0, stream>>>(Xpad, nxp);
        long nyp = (long)NB * TP * CCH;
        zero_bf16<<<cdiv((int)nyp, 256), 256, 0, stream>>>(Y0pad, nyp);
        long nx = (long)NB * CH * TT;
        transpose_in<<<cdiv((int)nx, 256), 256, 0, stream>>>(x, mask, Xf, Xbf);
    }

    const float scale = 1.0f / sqrtf((float)DKH);

    for (int i = 0; i < LYR; ++i) {
        const bf16_t* wq = Wqb + (long)i * CH * CH;
        const bf16_t* wk = Wkb + (long)i * CH * CH;
        const bf16_t* wv = Wvb + (long)i * CH * CH;
        const bf16_t* wo = Wob + (long)i * CH * CH;
        const bf16_t* w0 = W0p + (long)i * CCH * KS * CH;
        const bf16_t* w1 = W1p + (long)i * CH * KS * CCH;

        // Q (scale folded), K: token-major bf16.  V: channel-major bf16.
        gemm(wq, CH, 0, Xbf, CH, 0, nullptr, 0, Qbf, 0, CH,
             CH, NT, CH, 1, bq + i * CH, scale, F_STORE_T | F_BIAS | F_OUT_BF16, nullptr, 0, 1);
        gemm(wk, CH, 0, Xbf, CH, 0, nullptr, 0, Kbf, 0, CH,
             CH, NT, CH, 1, bk + i * CH, 1.0f, F_STORE_T | F_BIAS | F_OUT_BF16, nullptr, 0, 1);
        gemm(wv, CH, 0, Xbf, CH, 0, nullptr, 0, Vbf, 0, NT,
             CH, NT, CH, 1, bv + i * CH, 1.0f, F_BIAS | F_OUT_BF16, nullptr, 0, 1);

        // attention, chunked per batch to bound scratch
        for (int b = 0; b < NB; ++b) {
            // S[h,t,s] = qs . k    (z = h)
            gemm(Qbf + (long)b * TT * CH, CH, DKH,
                 Kbf + (long)b * TT * CH, CH, DKH,
                 S, (long)TT * TT, nullptr, 0, TT,
                 TT, TT, DKH, 1, nullptr, 1.0f, F_OUT_F32, nullptr, 0, NH);
            softmax_rel<<<dim3(TT, NH), 128, 0, stream>>>(
                S, P, Qbf, relk + (long)i * (2 * RW + 1) * DKH, mask, b);
            // out[t,d] = P x V   (z = h)
            gemm(P, TT, (long)TT * TT,
                 Vbf + (long)b * TT, NT, (long)DKH * NT,
                 attnF + (long)b * TT * CH, DKH, nullptr, 0, CH,
                 TT, DKH, TT, 1, nullptr, 1.0f, F_OUT_F32, nullptr, 0, NH);
            relv_add<<<dim3(TT, NH), DKH, 0, stream>>>(
                P, relv + (long)i * (2 * RW + 1) * DKH, attnF, b);
        }

        // O projection + LN0
        {
            long n = (long)NT * CH;
            cvt_f32_bf16<<<cdiv((int)n, 256), 256, 0, stream>>>(attnF, attnB, n);
        }
        gemm(wo, CH, 0, attnB, CH, 0, tmpF, 0, nullptr, 0, CH,
             CH, NT, CH, 1, bo + i * CH, 1.0f, F_STORE_T | F_BIAS | F_OUT_F32, nullptr, 0, 1);
        ln_kernel<<<NT, 128, 0, stream>>>(tmpF, Xf, ln0g + i * CH, ln0b + i * CH,
                                          mask, 0, newX, 0, nullptr, Xpad, 1);

        // conv FFN: conv0 (3-tap, ReLU, mask, bf16 padded out), conv1 (3-tap)
        gemm(w0, KS * CH, 0,
             Xpad, CH, (long)TP * CH,
             nullptr, 0, Y0pad + CCH, (long)TP * CCH, CCH,
             CCH, TT, CH, KS, c0b + i * CCH, 1.0f,
             F_STORE_T | F_BIAS | F_RELU | F_OUT_BF16 | F_MASKN, mask, TT, NB);
        gemm(w1, KS * CCH, 0,
             Y0pad, CCH, (long)TP * CCH,
             Y1, (long)TT * CH, nullptr, 0, CH,
             CH, TT, CCH, KS, c1b + i * CH, 1.0f,
             F_STORE_T | F_BIAS | F_OUT_F32, nullptr, 0, NB);
        ln_kernel<<<NT, 128, 0, stream>>>(Y1, newX, ln1g + i * CH, ln1b + i * CH,
                                          mask, 1, Xf, 1, Xbf, nullptr, 1);
    }

    {
        long n = (long)NB * CH * TT;
        transpose_out<<<cdiv((int)n, 256), 256, 0, stream>>>(Xf, mask, (float*)d_out);
    }
}
